// DecoderRNN_74105365725633
// MI455X (gfx1250) — matrix operand order repaired and checked by reading, not compiled, not run on hardware
//
#include <hip/hip_runtime.h>
#include <hip/hip_bf16.h>
#include <math.h>

typedef __attribute__((ext_vector_type(16))) __bf16 v16bf;
typedef __attribute__((ext_vector_type(8)))  float  v8f;

#define HDIM 1024
#define BATCH 64
#define VOCAB 50257
#define TSTEPS 10

__device__ __forceinline__ float sigmoidf_(float x) {
    return 1.0f / (1.0f + __expf(-x));
}

// Load A fragment (16x32, bf16) from row-major f32 matrix A[*, K].
// lane m=l&15; kh=(l>>4)*16; element e -> K=kh+e (contiguous, matches B)
__device__ __forceinline__ v16bf load_a_frag(const float* __restrict__ A, int K,
                                             int mbase, int kb, int lane) {
    int m  = mbase + (lane & 15);
    int kh = (lane >> 4) * 16;
    const float* p = A + (size_t)m * K + kb + kh;
    v16bf a;
#pragma unroll
    for (int e = 0; e < 8; ++e) a[e] = (__bf16)p[e];
#pragma unroll
    for (int e = 0; e < 8; ++e) a[8 + e] = (__bf16)p[8 + e];
    return a;
}

// Load B fragment (32x16, bf16) where B[k][n] = W[n][k], W row-major [N,K].
// this code's order: lane n=l&15; koff=(l>>4)*16; element e -> K=koff+e (contig row read)
// Caller guarantees n_global is in-bounds (clamped); no divergence here.
__device__ __forceinline__ v16bf load_b_frag(const float* __restrict__ W, int K,
                                             int n_global, int kb, int lane) {
    int koff = (lane >> 4) * 16;
    const float* p = W + (size_t)n_global * K + kb + koff;
    v16bf b;
#pragma unroll
    for (int e = 0; e < 16; ++e) b[e] = (__bf16)p[e];
    return b;
}

// gates[64,4096] = x @ w_ih^T + h @ w_hh^T + b_ih + b_hh
// grid: (64 n-groups of 4 tiles, 4 m-tiles), block = 32 (one wave -> 16x64 strip)
__global__ void gates_kernel(const float* __restrict__ x, const float* __restrict__ h,
                             const float* __restrict__ w_ih, const float* __restrict__ w_hh,
                             const float* __restrict__ b_ih, const float* __restrict__ b_hh,
                             float* __restrict__ gates) {
    int lane  = threadIdx.x;
    int mtile = blockIdx.y;
    int n0 = blockIdx.x * 64 + (lane & 15);
    v8f acc[4] = {};
    for (int kb = 0; kb < HDIM; kb += 32) {
        v16bf a = load_a_frag(x, HDIM, mtile * 16, kb, lane);
#pragma unroll
        for (int j = 0; j < 4; ++j) {
            v16bf b = load_b_frag(w_ih, HDIM, n0 + 16 * j, kb, lane);
            acc[j] = __builtin_amdgcn_wmma_f32_16x16x32_bf16(false, a, false, b,
                                                             (short)0, acc[j], false, false);
        }
    }
    for (int kb = 0; kb < HDIM; kb += 32) {
        v16bf a = load_a_frag(h, HDIM, mtile * 16, kb, lane);
#pragma unroll
        for (int j = 0; j < 4; ++j) {
            v16bf b = load_b_frag(w_hh, HDIM, n0 + 16 * j, kb, lane);
            acc[j] = __builtin_amdgcn_wmma_f32_16x16x32_bf16(false, a, false, b,
                                                             (short)0, acc[j], false, false);
        }
    }
    int mrow = mtile * 16 + 8 * (lane >> 4);
#pragma unroll
    for (int j = 0; j < 4; ++j) {
        int n = n0 + 16 * j;
        float bias = b_ih[n] + b_hh[n];
#pragma unroll
        for (int r = 0; r < 8; ++r)
            gates[(size_t)(mrow + r) * (4 * HDIM) + n] = acc[j][r] + bias;
    }
}

// logits[64,V] = h1 @ w_out^T + b_out
// grid: (786 n-groups of 4 tiles, 4 m-tiles), block=32. Ragged edge handled by
// clamping the weight row (dead columns compute garbage) and guarding stores.
__global__ void logits_kernel(const float* __restrict__ h1, const float* __restrict__ w_out,
                              const float* __restrict__ b_out, float* __restrict__ logits) {
    int lane  = threadIdx.x;
    int mtile = blockIdx.y;
    int n0 = blockIdx.x * 64 + (lane & 15);
    int nc[4];
#pragma unroll
    for (int j = 0; j < 4; ++j) {
        int n = n0 + 16 * j;
        nc[j] = (n < VOCAB) ? n : (VOCAB - 1);  // clamp: loads stay in-bounds
    }
    v8f acc[4] = {};
    for (int kb = 0; kb < HDIM; kb += 32) {
        v16bf a = load_a_frag(h1, HDIM, mtile * 16, kb, lane);
#pragma unroll
        for (int j = 0; j < 4; ++j) {
            v16bf b = load_b_frag(w_out, HDIM, nc[j], kb, lane);
            acc[j] = __builtin_amdgcn_wmma_f32_16x16x32_bf16(false, a, false, b,
                                                             (short)0, acc[j], false, false);
        }
    }
    int mrow = mtile * 16 + 8 * (lane >> 4);
#pragma unroll
    for (int j = 0; j < 4; ++j) {
        int n = n0 + 16 * j;
        if (n < VOCAB) {
            float bias = b_out[n];
#pragma unroll
            for (int r = 0; r < 8; ++r)
                logits[(size_t)(mrow + r) * VOCAB + n] = acc[j][r] + bias;
        }
    }
}

// Elementwise LSTM cell update: gate order i,f,g,o
__global__ void lstm_pointwise(const float* __restrict__ gates,
                               float* __restrict__ h, float* __restrict__ c) {
    int idx = blockIdx.x * blockDim.x + threadIdx.x;
    if (idx >= BATCH * HDIM) return;
    int b = idx >> 10, j = idx & (HDIM - 1);
    const float* g = gates + (size_t)b * (4 * HDIM);
    float ig = sigmoidf_(g[j]);
    float fg = sigmoidf_(g[HDIM + j]);
    float gg = tanhf(g[2 * HDIM + j]);
    float og = sigmoidf_(g[3 * HDIM + j]);
    float cn = fg * c[idx] + ig * gg;
    c[idx] = cn;
    h[idx] = og * tanhf(cn);
}

__global__ void embed_kernel(const int* __restrict__ tok, const float* __restrict__ emb,
                             float* __restrict__ x) {
    int idx = blockIdx.x * blockDim.x + threadIdx.x;
    if (idx >= BATCH * HDIM) return;
    int b = idx >> 10, j = idx & (HDIM - 1);
    x[idx] = emb[(size_t)tok[b] * HDIM + j];
}

__global__ void init_kernel(const float* __restrict__ h0, const float* __restrict__ c0,
                            float* __restrict__ h0s, float* __restrict__ c0s,
                            float* __restrict__ h1s, float* __restrict__ c1s,
                            int* __restrict__ tok) {
    int idx = blockIdx.x * blockDim.x + threadIdx.x;
    if (idx >= BATCH * HDIM) return;
    h0s[idx] = h0[idx];
    c0s[idx] = c0[idx];
    h1s[idx] = h0[BATCH * HDIM + idx];
    c1s[idx] = c0[BATCH * HDIM + idx];
    if (idx < BATCH) tok[idx] = 1;  // SOS token
}

// One block per batch row: argmax (first-occurrence ties), log_softmax, write output
__global__ void softmax_argmax_kernel(const float* __restrict__ logits,
                                      float* __restrict__ out, int* __restrict__ tok, int t) {
    __shared__ float smax[256];
    __shared__ int   sidx[256];
    __shared__ float ssum[256];
    int b = blockIdx.x, tid = threadIdx.x;
    const float* row = logits + (size_t)b * VOCAB;

    float best = -INFINITY; int bi = 0x7fffffff;
    for (int v = tid; v < VOCAB; v += 256) {
        float val = row[v];
        if (val > best || (val == best && v < bi)) { best = val; bi = v; }
    }
    smax[tid] = best; sidx[tid] = bi;
    __syncthreads();
    for (int s = 128; s > 0; s >>= 1) {
        if (tid < s) {
            if (smax[tid + s] > smax[tid] ||
                (smax[tid + s] == smax[tid] && sidx[tid + s] < sidx[tid])) {
                smax[tid] = smax[tid + s];
                sidx[tid] = sidx[tid + s];
            }
        }
        __syncthreads();
    }
    float rowmax = smax[0];
    if (tid == 0) tok[b] = sidx[0];

    float lsum = 0.0f;
    for (int v = tid; v < VOCAB; v += 256) lsum += __expf(row[v] - rowmax);
    ssum[tid] = lsum;
    __syncthreads();
    for (int s = 128; s > 0; s >>= 1) {
        if (tid < s) ssum[tid] += ssum[tid + s];
        __syncthreads();
    }
    float logZ = rowmax + __logf(ssum[0]);

    float* orow = out + (size_t)b * TSTEPS * VOCAB + (size_t)t * VOCAB;
    for (int v = tid; v < VOCAB; v += 256) orow[v] = row[v] - logZ;
}

// Pack h_final [2,B,H] then c_final [2,B,H] after log_probs in d_out
__global__ void finalize_kernel(const float* __restrict__ h0s, const float* __restrict__ c0s,
                                const float* __restrict__ h1s, const float* __restrict__ c1s,
                                float* __restrict__ out) {
    int idx = blockIdx.x * blockDim.x + threadIdx.x;
    if (idx >= BATCH * HDIM) return;
    size_t base = (size_t)BATCH * TSTEPS * VOCAB;
    const size_t S = (size_t)BATCH * HDIM;
    out[base + idx]         = h0s[idx];
    out[base + S + idx]     = h1s[idx];
    out[base + 2 * S + idx] = c0s[idx];
    out[base + 3 * S + idx] = c1s[idx];
}

extern "C" void kernel_launch(void* const* d_in, const int* in_sizes, int n_in,
                              void* d_out, int out_size, void* d_ws, size_t ws_size,
                              hipStream_t stream) {
    (void)in_sizes; (void)n_in; (void)out_size; (void)ws_size;
    // Input order per setup_inputs(): encoder_output(unused), h0, c0, emb,
    // w_ih0, w_hh0, b_ih0, b_hh0, w_ih1, w_hh1, b_ih1, b_hh1, w_out, b_out
    const float* h0    = (const float*)d_in[1];
    const float* c0    = (const float*)d_in[2];
    const float* emb   = (const float*)d_in[3];
    const float* w_ih0 = (const float*)d_in[4];
    const float* w_hh0 = (const float*)d_in[5];
    const float* b_ih0 = (const float*)d_in[6];
    const float* b_hh0 = (const float*)d_in[7];
    const float* w_ih1 = (const float*)d_in[8];
    const float* w_hh1 = (const float*)d_in[9];
    const float* b_ih1 = (const float*)d_in[10];
    const float* b_hh1 = (const float*)d_in[11];
    const float* w_out = (const float*)d_in[12];
    const float* b_out = (const float*)d_in[13];
    float* out = (float*)d_out;

    // Workspace layout (floats); first 64 ints are the token buffer.
    int*   tok    = (int*)d_ws;
    float* wsf    = (float*)d_ws;
    const size_t S  = (size_t)BATCH * HDIM;      // 65536
    float* x      = wsf + 64;
    float* gates  = x + S;                        // 64 x 4096
    float* h0s    = gates + 4 * S;
    float* c0s    = h0s + S;
    float* h1s    = c0s + S;
    float* c1s    = h1s + S;
    float* logits = c1s + S;                      // 64 x 50257

    const int NG_G = (4 * HDIM) / 64;             // 64 groups of 4 n-tiles
    const int NG_L = (VOCAB + 63) / 64;           // 786 groups (covers 50304 cols)
    dim3 gGates(NG_G, 4), gLogits(NG_L, 4);

    init_kernel<<<(BATCH * HDIM + 255) / 256, 256, 0, stream>>>(h0, c0, h0s, c0s, h1s, c1s, tok);

    for (int t = 0; t < TSTEPS; ++t) {
        embed_kernel<<<(BATCH * HDIM + 255) / 256, 256, 0, stream>>>(tok, emb, x);

        gates_kernel<<<gGates, 32, 0, stream>>>(x, h0s, w_ih0, w_hh0, b_ih0, b_hh0, gates);
        lstm_pointwise<<<(BATCH * HDIM + 255) / 256, 256, 0, stream>>>(gates, h0s, c0s);

        gates_kernel<<<gGates, 32, 0, stream>>>(h0s, h1s, w_ih1, w_hh1, b_ih1, b_hh1, gates);
        lstm_pointwise<<<(BATCH * HDIM + 255) / 256, 256, 0, stream>>>(gates, h1s, c1s);

        logits_kernel<<<gLogits, 32, 0, stream>>>(h1s, w_out, b_out, logits);
        softmax_argmax_kernel<<<BATCH, 256, 0, stream>>>(logits, out, tok, t);
    }

    finalize_kernel<<<(BATCH * HDIM + 255) / 256, 256, 0, stream>>>(h0s, c0s, h1s, c1s, out);
}
